// Net_coor_51187420233847
// MI455X (gfx1250) — matrix operand order repaired and checked by reading, not compiled, not run on hardware
//
#include <hip/hip_runtime.h>
#include <math.h>

typedef __bf16  v16bf __attribute__((ext_vector_type(16)));
typedef float   v8f   __attribute__((ext_vector_type(8)));

#define LDSK 136   // LDS row stride in halves: 128 + 8 pad (keeps 16B alignment: 136 = 17*8)

union bf16x16 { v16bf v; uint4 q[2]; };
union bf16x4  { __bf16 h[4]; uint2 u; };

// A fragment: halves p[0..7] and p[16..23] (ISA 16x32 bf16 A layout per lane half)
__device__ __forceinline__ v16bf ld_frag_a(const __bf16* p)
{
    bf16x16 u;
    u.q[0] = *(const uint4*)(p);
    u.q[1] = *(const uint4*)(p + 16);
    return u.v;
}
// B fragment: halves p[0..7] and p[16..23] (same K order as A)
__device__ __forceinline__ v16bf ld_frag_b(const __bf16* p)
{
    bf16x16 u;
    u.q[0] = *(const uint4*)(p);
    u.q[1] = *(const uint4*)(p + 16);
    return u.v;
}

// ---------------------------------------------------------------------------
// fp32 -> bf16 cast, 4 elements/thread, vector loads/stores
// ---------------------------------------------------------------------------
__global__ void cast_bf16_kernel(const float* __restrict__ in,
                                 __bf16* __restrict__ out, long n4)
{
    long i = (long)blockIdx.x * blockDim.x + threadIdx.x;
    if (i >= n4) return;
    float4 f = ((const float4*)in)[i];
    bf16x4 o;
    o.h[0] = (__bf16)f.x; o.h[1] = (__bf16)f.y;
    o.h[2] = (__bf16)f.z; o.h[3] = (__bf16)f.w;
    ((uint2*)out)[i] = o.u;
}

// ---------------------------------------------------------------------------
// GEMM: Y[M x C] = Xb[M x K](bf16) @ Wb[C x K](bf16)^T + bias, fp32 out.
// Block = 256 threads (8 waves). Tile = 128 rows x 32 cols, full K panel in
// LDS. Each wave: one 16-row strip, two 16x16 accumulators (32 cols),
// 2 WMMA per 32-K step. K in {64,128} (multiple of 32).
// ---------------------------------------------------------------------------
__global__ void gemm_bf16_wmma(const __bf16* __restrict__ Xb,
                               const __bf16* __restrict__ Wb,
                               const float* __restrict__ bias,
                               float* __restrict__ Y, int ldy,
                               int M, int K, int C)
{
    __shared__ __bf16 xs[128 * LDSK];   // 34816 B
    __shared__ __bf16 wt[32 * LDSK];    //  8704 B

    const int tid  = threadIdx.x;
    const int wave = tid >> 5;
    const int lane = tid & 31;
    const int m0   = blockIdx.x * 128;
    const int c0   = blockIdx.y * 32;
    const int kc8  = K >> 3;            // 8-half (16B) chunks per row

    // Stage X panel: 128 x K halves, 16B vector chunks
    for (int i = tid; i < 128 * kc8; i += 256) {
        int r  = i / kc8;
        int ck = i - r * kc8;
        uint4 val = {0u, 0u, 0u, 0u};
        int gr = m0 + r;
        if (gr < M) val = ((const uint4*)Xb)[(long)gr * kc8 + ck];
        *(uint4*)&xs[r * LDSK + (ck << 3)] = val;
    }
    // Stage W panel: 32 x K halves (rows = output channels c0..c0+31)
    for (int i = tid; i < 32 * kc8; i += 256) {
        int n  = i / kc8;
        int ck = i - n * kc8;
        uint4 val = {0u, 0u, 0u, 0u};
        int gc = c0 + n;
        if (gc < C) val = ((const uint4*)Wb)[(long)gc * kc8 + ck];
        *(uint4*)&wt[n * LDSK + (ck << 3)] = val;
    }
    __syncthreads();

    v8f acc0 = {}, acc1 = {};
    const int rowA = (wave << 4) + (lane & 15);
    const int kbA  = (lane < 16) ? 0 : 8;    // A: K base per lane half
    const int kbB  = (lane < 16) ? 0 : 8;    // B: K base per lane half
    const __bf16* xrow  = &xs[rowA * LDSK + kbA];
    const __bf16* wrow0 = &wt[(lane & 15) * LDSK + kbB];
    const __bf16* wrow1 = &wt[(16 + (lane & 15)) * LDSK + kbB];

#pragma unroll 4
    for (int k0 = 0; k0 < K; k0 += 32) {
        v16bf a  = ld_frag_a(xrow + k0);
        v16bf b0 = ld_frag_b(wrow0 + k0);
        v16bf b1 = ld_frag_b(wrow1 + k0);
        acc0 = __builtin_amdgcn_wmma_f32_16x16x32_bf16(
            false, a, false, b0, (short)0, acc0, false, false);
        acc1 = __builtin_amdgcn_wmma_f32_16x16x32_bf16(
            false, a, false, b1, (short)0, acc1, false, false);
    }

    // D layout: VGPR r -> M = r + 8*(lane>=16); N = lane&15
    const int rowO = m0 + (wave << 4) + ((lane >> 4) << 3);
    const int colL = lane & 15;
#pragma unroll
    for (int r = 0; r < 8; ++r) {
        int row = rowO + r;
        if (row >= M) continue;
        int col0 = c0 + colL;
        int col1 = c0 + 16 + colL;
        if (col0 < C) Y[(long)row * ldy + col0] = acc0[r] + (bias ? bias[col0] : 0.0f);
        if (col1 < C) Y[(long)row * ldy + col1] = acc1[r] + (bias ? bias[col1] : 0.0f);
    }
}

// ---------------------------------------------------------------------------
// Edge kernels: one wave per edge; We cached in LDS; e = edge_attr @ We^T
// recomputed on the fly (saves an E x 128 fp32 buffer, ~800 MB HBM/layer).
// ---------------------------------------------------------------------------
__global__ void edge_alpha_kernel(const float* __restrict__ q,
                                  const float* __restrict__ kb,
                                  const float* __restrict__ eattr,
                                  const float* __restrict__ We,
                                  const int* __restrict__ src,
                                  const int* __restrict__ dst,
                                  float* __restrict__ alpha,
                                  int E, int C, int ld, float scale)
{
    __shared__ float we_s[128 * 16];
    const int tid = threadIdx.x;
    for (int i = tid; i < C * 16; i += blockDim.x) we_s[i] = We[i];
    __syncthreads();

    const int wave = tid >> 5, lane = tid & 31;
    const int e = blockIdx.x * (blockDim.x >> 5) + wave;
    if (e >= E) return;

    const int si = src[e], di = dst[e];
    float ea[16];
#pragma unroll
    for (int t = 0; t < 16; ++t) ea[t] = eattr[(long)e * 16 + t];

    float part = 0.0f;
    for (int c = lane; c < C; c += 32) {
        float ec = 0.0f;
#pragma unroll
        for (int t = 0; t < 16; ++t) ec += we_s[c * 16 + t] * ea[t];
        part += q[(long)di * ld + c] * (kb[(long)si * ld + c] + ec);
    }
#pragma unroll
    for (int off = 16; off > 0; off >>= 1) part += __shfl_xor(part, off, 32);
    if (lane == 0) alpha[e] = part * scale;
}

__global__ void seg_max_kernel(const float* __restrict__ alpha,
                               const int* __restrict__ dst,
                               float* __restrict__ m, int E)
{
    int e = blockIdx.x * blockDim.x + threadIdx.x;
    if (e >= E) return;
    float val = alpha[e];
    unsigned int* addr = (unsigned int*)&m[dst[e]];
    unsigned int old = *addr;
    while (true) {
        float cur = __uint_as_float(old);
        if (cur >= val) break;
        unsigned int assumed = old;
        old = atomicCAS(addr, assumed, __float_as_uint(val));
        if (old == assumed) break;
    }
}

__global__ void fix_m_kernel(float* __restrict__ m, int N)
{
    int i = blockIdx.x * blockDim.x + threadIdx.x;
    if (i >= N) return;
    float v = m[i];
    if (!isfinite(v)) m[i] = 0.0f;   // nodes with no in-edges
}

__global__ void edge_exp_kernel(float* __restrict__ alpha,
                                const int* __restrict__ dst,
                                const float* __restrict__ m,
                                float* __restrict__ s, int E)
{
    int e = blockIdx.x * blockDim.x + threadIdx.x;
    if (e >= E) return;
    int di = dst[e];
    float ex = __expf(alpha[e] - m[di]);
    alpha[e] = ex;
    atomicAdd(&s[di], ex);
}

__global__ void edge_msg_kernel(const float* __restrict__ vb,
                                const float* __restrict__ eattr,
                                const float* __restrict__ We,
                                const int* __restrict__ src,
                                const int* __restrict__ dst,
                                const float* __restrict__ ex,
                                const float* __restrict__ s,
                                float* __restrict__ agg,
                                int E, int C, int ld)
{
    __shared__ float we_s[128 * 16];
    const int tid = threadIdx.x;
    for (int i = tid; i < C * 16; i += blockDim.x) we_s[i] = We[i];
    __syncthreads();

    const int wave = tid >> 5, lane = tid & 31;
    const int e = blockIdx.x * (blockDim.x >> 5) + wave;
    if (e >= E) return;

    const int si = src[e], di = dst[e];
    const float a = ex[e] / (s[di] + 1e-16f);

    float ea[16];
#pragma unroll
    for (int t = 0; t < 16; ++t) ea[t] = eattr[(long)e * 16 + t];

    for (int c = lane; c < C; c += 32) {
        float ec = 0.0f;
#pragma unroll
        for (int t = 0; t < 16; ++t) ec += we_s[c * 16 + t] * ea[t];
        atomicAdd(&agg[(long)di * ld + c], (vb[(long)si * ld + c] + ec) * a);
    }
}

// out = agg + skip; mode 1: gelu; mode 2: gelu + residual(hprev); mode 0: plain
__global__ void combine_kernel(const float* __restrict__ agg,
                               const float* __restrict__ skp,
                               const float* __restrict__ hprev,
                               float* __restrict__ out,
                               int N, int C, int ld_in, int ld_out, int mode)
{
    long idx = (long)blockIdx.x * blockDim.x + threadIdx.x;
    long total = (long)N * C;
    if (idx >= total) return;
    int r = (int)(idx / C), c = (int)(idx % C);
    float xv = agg[(long)r * ld_in + c] + skp[(long)r * ld_in + c];
    if (mode >= 1)
        xv = 0.5f * xv * (1.0f + erff(xv * 0.70710678118654752f)); // exact GELU
    if (mode == 2)
        xv += hprev[(long)r * ld_in + c];
    out[(long)r * ld_out + c] = xv;
}

__global__ void fill_kernel(float* __restrict__ p, float val, long n)
{
    long i = (long)blockIdx.x * blockDim.x + threadIdx.x;
    if (i < n) p[i] = val;
}

// ---------------------------------------------------------------------------
// Host orchestration
// ---------------------------------------------------------------------------
extern "C" void kernel_launch(void* const* d_in, const int* in_sizes, int n_in,
                              void* d_out, int out_size, void* d_ws, size_t ws_size,
                              hipStream_t stream)
{
    (void)n_in; (void)out_size; (void)ws_size;

    const float* x     = (const float*)d_in[0];
    const int*   ei    = (const int*)d_in[1];
    const float* eattr = (const float*)d_in[2];

    const int N = in_sizes[0] / 64;   // F_IN = 64
    const int E = in_sizes[1] / 2;
    const int* src = ei;
    const int* dst = ei + E;

    const float* p1[9]; for (int i = 0; i < 9; ++i) p1[i] = (const float*)d_in[3 + i];
    const float* pm[9]; for (int i = 0; i < 9; ++i) pm[i] = (const float*)d_in[12 + i];
    const float* pl[9]; for (int i = 0; i < 9; ++i) pl[i] = (const float*)d_in[21 + i];
    const int L = in_sizes[12] / (128 * 128);   // stacked middle layers

    // Workspace carve-up: fp32 region then bf16 region (all 16B aligned)
    float* ws = (float*)d_ws;
    const size_t NB = (size_t)N * 128;
    float* h     = ws;
    float* q     = ws + 1 * NB;
    float* kbuf  = ws + 2 * NB;
    float* vbuf  = ws + 3 * NB;
    float* agg   = ws + 4 * NB;
    float* skp   = ws + 5 * NB;
    float* hn    = ws + 6 * NB;
    float* alpha = ws + 7 * NB;          // E floats
    float* mbuf  = alpha + E;            // N floats
    float* sbuf  = mbuf + N;             // N floats
    __bf16* xb  = (__bf16*)(sbuf + N);   // N*128 halves (max)
    __bf16* wqb = xb  + NB;              // 128*128 halves each
    __bf16* wkb = wqb + 128 * 128;
    __bf16* wvb = wkb + 128 * 128;
    __bf16* wsb = wvb + 128 * 128;

    const int TB = 256;
    const int edgeWaveBlocks = (E + 7) / 8;      // 8 waves per 256-thread block
    const int edgeThrBlocks  = (E + TB - 1) / TB;
    const int nodeBlocks     = (N + TB - 1) / TB;

    auto cast = [&](const float* in, __bf16* out, long n) {
        long n4 = n >> 2;
        cast_bf16_kernel<<<(int)((n4 + TB - 1) / TB), TB, 0, stream>>>(in, out, n4);
    };

    auto run_layer = [&](const float* Xin, int K,
                         const float* Wq, const float* bq,
                         const float* Wk, const float* bk,
                         const float* Wv, const float* bv,
                         const float* We,
                         const float* Wsk, const float* bsk,
                         int C, int ld,
                         float* outbuf, int ld_out, int mode, const float* hprev)
    {
        // bf16 staging for the matrix engine
        cast(Xin, xb,  (long)N * K);
        cast(Wq,  wqb, (long)C * K);
        cast(Wk,  wkb, (long)C * K);
        cast(Wv,  wvb, (long)C * K);
        cast(Wsk, wsb, (long)C * K);

        dim3 gg((N + 127) / 128, (C + 31) / 32);
        gemm_bf16_wmma<<<gg, TB, 0, stream>>>(xb, wqb, bq,  q,    ld, N, K, C);
        gemm_bf16_wmma<<<gg, TB, 0, stream>>>(xb, wkb, bk,  kbuf, ld, N, K, C);
        gemm_bf16_wmma<<<gg, TB, 0, stream>>>(xb, wvb, bv,  vbuf, ld, N, K, C);
        gemm_bf16_wmma<<<gg, TB, 0, stream>>>(xb, wsb, bsk, skp,  ld, N, K, C);

        long aggN = (long)N * ld;
        fill_kernel<<<(int)((aggN + TB - 1) / TB), TB, 0, stream>>>(agg, 0.0f, aggN);
        fill_kernel<<<nodeBlocks, TB, 0, stream>>>(sbuf, 0.0f, N);
        fill_kernel<<<nodeBlocks, TB, 0, stream>>>(mbuf, -INFINITY, N);

        float scale = 1.0f / sqrtf((float)C);
        edge_alpha_kernel<<<edgeWaveBlocks, TB, 0, stream>>>(
            q, kbuf, eattr, We, src, dst, alpha, E, C, ld, scale);
        seg_max_kernel<<<edgeThrBlocks, TB, 0, stream>>>(alpha, dst, mbuf, E);
        fix_m_kernel<<<nodeBlocks, TB, 0, stream>>>(mbuf, N);
        edge_exp_kernel<<<edgeThrBlocks, TB, 0, stream>>>(alpha, dst, mbuf, sbuf, E);
        edge_msg_kernel<<<edgeWaveBlocks, TB, 0, stream>>>(
            vbuf, eattr, We, src, dst, alpha, sbuf, agg, E, C, ld);

        long total = (long)N * C;
        combine_kernel<<<(int)((total + TB - 1) / TB), TB, 0, stream>>>(
            agg, skp, hprev, outbuf, N, C, ld, ld_out, mode);
    };

    // conv1: F_IN=64 -> 128, GELU (no residual)
    run_layer(x, 64,
              p1[0], p1[1], p1[2], p1[3], p1[4], p1[5], p1[6], p1[7], p1[8],
              128, 128, h, 128, /*mode=*/1, nullptr);

    // L middle convs: 128 -> 128, GELU + residual
    float* cur = h;
    float* nxt = hn;
    for (int l = 0; l < L; ++l) {
        run_layer(cur, 128,
                  pm[0] + (long)l * 128 * 128, pm[1] + (long)l * 128,
                  pm[2] + (long)l * 128 * 128, pm[3] + (long)l * 128,
                  pm[4] + (long)l * 128 * 128, pm[5] + (long)l * 128,
                  pm[6] + (long)l * 128 * 16,
                  pm[7] + (long)l * 128 * 128, pm[8] + (long)l * 128,
                  128, 128, nxt, 128, /*mode=*/2, cur);
        float* t = cur; cur = nxt; nxt = t;
    }

    // convl: 128 -> 3 (padded to 32-wide tile), plain output
    run_layer(cur, 128,
              pl[0], pl[1], pl[2], pl[3], pl[4], pl[5], pl[6], pl[7], pl[8],
              3, 16, (float*)d_out, 3, /*mode=*/0, nullptr);
}
